// GraphDenselyConnectedTransformerLayer_2070174236918
// MI455X (gfx1250) — compile-verified, matrix operand order checked by reading, not run on hardware
//
#include <hip/hip_runtime.h>
#include <hip/hip_bf16.h>
#include <math.h>

// ---------------------------------------------------------------------------
// Problem constants (from reference)
// ---------------------------------------------------------------------------
#define V_NODES 16384
#define GRP     32
#define D_MODEL 256
#define N_HEADS 8
#define H_DIM   32
#define QKV_DIM 768          // 3*N*H
#define I_DIM   1024
#define LN_EPS  1e-12f

typedef __attribute__((ext_vector_type(8)))  _Float16 v8h;
typedef __attribute__((ext_vector_type(16))) _Float16 v16h;
typedef __attribute__((ext_vector_type(8)))  float    v8f;

// ---------------------------------------------------------------------------
// Fragment loader: per-lane 16 halves of one matrix row (row-major, ld = K).
// CDNA5 16-bit A 16x32 layout: lanes 0-15 (kb=0) hold K {0..7,16..23},
// lanes 16-31 (kb=8) hold K {8..15,24..31}.  Two contiguous 16B loads.
// ---------------------------------------------------------------------------
__device__ __forceinline__ v16h load_frag(const _Float16* __restrict__ row, int kb) {
    v8h x0 = *(const v8h*)(row + kb);
    v8h x1 = *(const v8h*)(row + kb + 16);
    v16h r;
#pragma unroll
    for (int i = 0; i < 8; ++i) { r[i] = x0[i]; r[i + 8] = x1[i]; }
    return r;
}

__device__ __forceinline__ v8f wmma_f16(v16h a, v16h b, v8f c) {
    return __builtin_amdgcn_wmma_f32_16x16x32_f16(
        /*neg_a=*/false, a, /*neg_b=*/false, b,
        /*c_mod=*/(short)0, c, /*reuse_a=*/false, /*reuse_b=*/false);
}

// ---------------------------------------------------------------------------
// f16 WMMA GEMM:  C[M x N] = A[M x K] * BT[N x K]^T   (fp32 accumulate)
// Register-blocked: one wave computes a 16x64 tile (A frag reused across 4
// WMMAs), with K-dimension double buffering so loads overlap WMMA execution.
//   mode 0: store fp32 into Cf
//   mode 1: exact GeLU, store f16 into Ch   (fused FFN1 epilogue)
// Block = (32,8): 8 waves stacked in M.  Requires M%128==0, N%64==0, K%32==0.
// ---------------------------------------------------------------------------
__global__ void gemm_f16_wmma(const _Float16* __restrict__ A,
                              const _Float16* __restrict__ BT,
                              float* __restrict__ Cf,
                              _Float16* __restrict__ Ch,
                              int M, int N, int K, int mode) {
    const int lane = threadIdx.x;           // 0..31
    const int lo   = lane & 15;
    const int hi   = lane >> 4;
    const int tn   = blockIdx.x;                            // 64-wide N tile
    const int tm   = blockIdx.y * blockDim.y + threadIdx.y; // 16-wide M tile

    const _Float16* __restrict__ arow = A + (size_t)(tm * 16 + lo) * K;
    const _Float16* __restrict__ brow0 = BT + (size_t)(tn * 64 +  0 + lo) * K;
    const _Float16* __restrict__ brow1 = BT + (size_t)(tn * 64 + 16 + lo) * K;
    const _Float16* __restrict__ brow2 = BT + (size_t)(tn * 64 + 32 + lo) * K;
    const _Float16* __restrict__ brow3 = BT + (size_t)(tn * 64 + 48 + lo) * K;

    v8f acc0 = {}, acc1 = {}, acc2 = {}, acc3 = {};

    // prologue: fragments for k-step 0
    int kb = hi * 8;
    v16h a  = load_frag(arow,  kb);
    v16h b0 = load_frag(brow0, kb);
    v16h b1 = load_frag(brow1, kb);
    v16h b2 = load_frag(brow2, kb);
    v16h b3 = load_frag(brow3, kb);

    for (int kk = 32; kk < K; kk += 32) {
        // issue next k-step's loads first so their latency hides behind WMMA
        const int kbn = kk + hi * 8;
        __builtin_prefetch(arow  + kbn + 32, 0, 3);   // gfx1250 global_prefetch_b8
        __builtin_prefetch(brow0 + kbn + 32, 0, 3);
        v16h an  = load_frag(arow,  kbn);
        v16h bn0 = load_frag(brow0, kbn);
        v16h bn1 = load_frag(brow1, kbn);
        v16h bn2 = load_frag(brow2, kbn);
        v16h bn3 = load_frag(brow3, kbn);

        acc0 = wmma_f16(a, b0, acc0);
        acc1 = wmma_f16(a, b1, acc1);
        acc2 = wmma_f16(a, b2, acc2);
        acc3 = wmma_f16(a, b3, acc3);

        a = an; b0 = bn0; b1 = bn1; b2 = bn2; b3 = bn3;
    }
    // epilogue k-step
    acc0 = wmma_f16(a, b0, acc0);
    acc1 = wmma_f16(a, b1, acc1);
    acc2 = wmma_f16(a, b2, acc2);
    acc3 = wmma_f16(a, b3, acc3);

    // C layout: VGPR r -> row = tm*16 + r + 8*hi, col tile c -> tn*64+c*16+lo
    const int crow0 = tm * 16 + hi * 8;
    const int ccol  = tn * 64 + lo;
    if (mode == 0) {
#pragma unroll
        for (int r = 0; r < 8; ++r) {
            const size_t rowoff = (size_t)(crow0 + r) * N + ccol;
            Cf[rowoff +  0] = acc0[r];
            Cf[rowoff + 16] = acc1[r];
            Cf[rowoff + 32] = acc2[r];
            Cf[rowoff + 48] = acc3[r];
        }
    } else {
#pragma unroll
        for (int r = 0; r < 8; ++r) {
            const size_t rowoff = (size_t)(crow0 + r) * N + ccol;
            float x0 = acc0[r], x1 = acc1[r], x2 = acc2[r], x3 = acc3[r];
            Ch[rowoff +  0] = (_Float16)(0.5f * x0 * (1.0f + erff(x0 * 0.70710678118654752f)));
            Ch[rowoff + 16] = (_Float16)(0.5f * x1 * (1.0f + erff(x1 * 0.70710678118654752f)));
            Ch[rowoff + 32] = (_Float16)(0.5f * x2 * (1.0f + erff(x2 * 0.70710678118654752f)));
            Ch[rowoff + 48] = (_Float16)(0.5f * x3 * (1.0f + erff(x3 * 0.70710678118654752f)));
        }
    }
}

// ---------------------------------------------------------------------------
// Block-dense attention: one wave per (head, group-of-32).
// qkv is fp32 (V x 768) laid out [node][head][q32|k32|v32].
// Output att (V x 256, head-major cols) stored as f16 for the W_o GEMM.
// ---------------------------------------------------------------------------
__global__ void attn_kernel(const float* __restrict__ qkv,
                            _Float16* __restrict__ att_h) {
    const int h = blockIdx.x;   // head
    const int g = blockIdx.y;   // group
    const int j = threadIdx.x;  // 0..31 : dst node within group / out column

    __shared__ float qs[32][33];
    __shared__ float ks[32][33];
    __shared__ float vs[32][33];
    __shared__ float ws[32][33];

    const size_t base = (size_t)(g * GRP + j) * QKV_DIM + h * (3 * H_DIM);
#pragma unroll
    for (int t = 0; t < H_DIM; t += 4) {
        float4 q4 = *(const float4*)(qkv + base + t);
        float4 k4 = *(const float4*)(qkv + base + H_DIM + t);
        float4 v4 = *(const float4*)(qkv + base + 2 * H_DIM + t);
        qs[j][t] = q4.x; qs[j][t+1] = q4.y; qs[j][t+2] = q4.z; qs[j][t+3] = q4.w;
        ks[j][t] = k4.x; ks[j][t+1] = k4.y; ks[j][t+2] = k4.z; ks[j][t+3] = k4.w;
        vs[j][t] = v4.x; vs[j][t+1] = v4.y; vs[j][t+2] = v4.z; vs[j][t+3] = v4.w;
    }
    __syncthreads();

    const float scale = 0.17677669529663689f;  // 1/sqrt(32)
    for (int i = 0; i < GRP; ++i) {
        float s = 0.0f;
#pragma unroll
        for (int t = 0; t < H_DIM; ++t) s += qs[i][t] * ks[j][t];
        s *= scale;
        float m = s;
#pragma unroll
        for (int off = 16; off; off >>= 1) m = fmaxf(m, __shfl_xor(m, off, 32));
        float e = __expf(s - m);
        float d = e;
#pragma unroll
        for (int off = 16; off; off >>= 1) d += __shfl_xor(d, off, 32);
        ws[i][j] = e / d;
    }
    __syncthreads();

    // lane j now owns output column t=j of the 32x32 (W @ V) product
    for (int i = 0; i < GRP; ++i) {
        float acc = 0.0f;
#pragma unroll
        for (int jj = 0; jj < GRP; ++jj) acc += ws[i][jj] * vs[jj][j];
        att_h[(size_t)(g * GRP + i) * D_MODEL + h * H_DIM + j] = (_Float16)acc;
    }
}

// ---------------------------------------------------------------------------
// Residual + LayerNorm over D=256.  One 256-thread block per row.
// Writes fp32 (always) and f16 (if outh != nullptr, feeds next WMMA GEMM).
// ---------------------------------------------------------------------------
__global__ void resid_ln_kernel(const float* __restrict__ x,
                                const float* __restrict__ res,
                                const float* __restrict__ gam,
                                const float* __restrict__ bet,
                                float* __restrict__ outf,
                                _Float16* __restrict__ outh) {
    const int r = blockIdx.x;
    const int t = threadIdx.x;   // 0..255
    __shared__ float red[256];

    const size_t idx = (size_t)r * D_MODEL + t;
    float v = x[idx] + res[idx];

    red[t] = v; __syncthreads();
#pragma unroll
    for (int s = 128; s; s >>= 1) { if (t < s) red[t] += red[t + s]; __syncthreads(); }
    float mu = red[0] * (1.0f / D_MODEL);
    __syncthreads();

    float dv = v - mu;
    red[t] = dv * dv; __syncthreads();
#pragma unroll
    for (int s = 128; s; s >>= 1) { if (t < s) red[t] += red[t + s]; __syncthreads(); }
    float var = red[0] * (1.0f / D_MODEL);

    float y = dv * rsqrtf(var + LN_EPS) * gam[t] + bet[t];
    outf[idx] = y;
    if (outh) outh[idx] = (_Float16)y;
}

// ---------------------------------------------------------------------------
// Precision casts
// ---------------------------------------------------------------------------
__global__ void cast_f16_kernel(const float* __restrict__ src,
                                _Float16* __restrict__ dst, int n) {
    int i = blockIdx.x * blockDim.x + threadIdx.x;
    const int stride = gridDim.x * blockDim.x;
    for (; i < n; i += stride) dst[i] = (_Float16)src[i];
}

// src is K x N row-major; dst is N x K (transposed) f16 -> BT for WMMA
__global__ void cast_transpose_f16_kernel(const float* __restrict__ src,
                                          _Float16* __restrict__ dst,
                                          int K, int N) {
    int i = blockIdx.x * blockDim.x + threadIdx.x;
    const int total = K * N;
    const int stride = gridDim.x * blockDim.x;
    for (; i < total; i += stride) {
        int k = i / N, n = i - k * N;
        dst[(size_t)n * K + k] = (_Float16)src[i];
    }
}

// ---------------------------------------------------------------------------
// Host-side launch sequence (graph-capture safe: only kernel launches)
// ---------------------------------------------------------------------------
extern "C" void kernel_launch(void* const* d_in, const int* in_sizes, int n_in,
                              void* d_out, int out_size, void* d_ws, size_t ws_size,
                              hipStream_t stream) {
    (void)in_sizes; (void)n_in; (void)out_size; (void)ws_size;

    const float* X      = (const float*)d_in[0];   // V x 256
    // d_in[1]: in_graph_node_pairs — structure is fixed dense 32-groups; implicit.
    const float* W_qkv  = (const float*)d_in[2];   // 256 x 768
    const float* W_o    = (const float*)d_in[3];   // 256 x 256
    const float* ln1_g  = (const float*)d_in[4];
    const float* ln1_b  = (const float*)d_in[5];
    const float* W_i    = (const float*)d_in[6];   // 256 x 1024
    const float* W_out2 = (const float*)d_in[7];   // 1024 x 256
    const float* ln2_g  = (const float*)d_in[8];
    const float* ln2_b  = (const float*)d_in[9];
    float* out = (float*)d_out;                    // V x 256 fp32

    // ---- workspace layout (byte offsets, lifetimes do not overlap) ----
    const size_t MB = 1ull << 20;
    char* ws = (char*)d_ws;
    float*    qkv_f32  = (float*)   (ws + 0);          // 48 MB   (steps 6-7)
    float*    tmp_f32  = (float*)   (ws + 0);          // 16 MB   (steps 8-9, qkv dead)
    _Float16* inter_h  = (_Float16*)(ws + 0);          // 32 MB   (steps 10-11, tmp dead)
    float*    tmp2_f32 = (float*)   (ws + 32 * MB);    // 16 MB   (steps 11-12)
    _Float16* Xh       = (_Float16*)(ws + 48 * MB);    //  8 MB   (steps 1-6)
    _Float16* att_h    = (_Float16*)(ws + 48 * MB);    //  8 MB   (steps 7-8, Xh dead)
    float*    pre_f32  = (float*)   (ws + 56 * MB);    // 16 MB   (steps 9-12)
    _Float16* pre_h    = (_Float16*)(ws + 72 * MB);    //  8 MB   (steps 9-10)
    _Float16* Wqkvt    = (_Float16*)(ws + 80 * MB);               // 768 x 256
    _Float16* Wot      = (_Float16*)(ws + 80 * MB + 393216);      // 256 x 256
    _Float16* Wit      = (_Float16*)(ws + 80 * MB + 524288);      // 1024 x 256
    _Float16* Wout2t   = (_Float16*)(ws + 80 * MB + 1048576);     // 256 x 1024

    // 1) casts (X) and weight cast+transposes
    cast_f16_kernel<<<4096, 256, 0, stream>>>(X, Xh, V_NODES * D_MODEL);
    cast_transpose_f16_kernel<<<768, 256, 0, stream>>>(W_qkv,  Wqkvt,  D_MODEL, QKV_DIM);
    cast_transpose_f16_kernel<<<256, 256, 0, stream>>>(W_o,    Wot,    D_MODEL, D_MODEL);
    cast_transpose_f16_kernel<<<1024, 256, 0, stream>>>(W_i,   Wit,    D_MODEL, I_DIM);
    cast_transpose_f16_kernel<<<1024, 256, 0, stream>>>(W_out2, Wout2t, I_DIM,  D_MODEL);

    const dim3 gblk(32, 8);

    // 6) qkv = Xh @ W_qkv           (16384 x 768, K=256)
    gemm_f16_wmma<<<dim3(QKV_DIM / 64, V_NODES / 128), gblk, 0, stream>>>(
        Xh, Wqkvt, qkv_f32, nullptr, V_NODES, QKV_DIM, D_MODEL, 0);

    // 7) block-dense attention -> att (f16)
    attn_kernel<<<dim3(N_HEADS, V_NODES / GRP), 32, 0, stream>>>(qkv_f32, att_h);

    // 8) tmp = att @ W_o            (16384 x 256, K=256)
    gemm_f16_wmma<<<dim3(D_MODEL / 64, V_NODES / 128), gblk, 0, stream>>>(
        att_h, Wot, tmp_f32, nullptr, V_NODES, D_MODEL, D_MODEL, 0);

    // 9) pre = LN(tmp + X)          (fp32 + f16 copies)
    resid_ln_kernel<<<V_NODES, D_MODEL, 0, stream>>>(
        tmp_f32, X, ln1_g, ln1_b, pre_f32, pre_h);

    // 10) inter = gelu(pre @ W_i)   (16384 x 1024, K=256) fused epilogue -> f16
    gemm_f16_wmma<<<dim3(I_DIM / 64, V_NODES / 128), gblk, 0, stream>>>(
        pre_h, Wit, nullptr, inter_h, V_NODES, I_DIM, D_MODEL, 1);

    // 11) tmp2 = inter @ W_out2     (16384 x 256, K=1024)
    gemm_f16_wmma<<<dim3(D_MODEL / 64, V_NODES / 128), gblk, 0, stream>>>(
        inter_h, Wout2t, tmp2_f32, nullptr, V_NODES, D_MODEL, I_DIM, 0);

    // 12) out = LN(tmp2 + pre)
    resid_ln_kernel<<<V_NODES, D_MODEL, 0, stream>>>(
        tmp2_f32, pre_f32, ln2_g, ln2_b, out, nullptr);
}
